// LightGCN_Cooccur_21930103013810
// MI455X (gfx1250) — matrix operand order repaired and checked by reading, not compiled, not run on hardware
//
#include <hip/hip_runtime.h>
#include <math.h>

typedef __attribute__((ext_vector_type(16))) _Float16 v16h;
typedef __attribute__((ext_vector_type(8)))  float    v8f;

namespace {
constexpr int kNU = 100000;
constexpr int kNI = 50000;
constexpr int kD  = 64;
constexpr int kL  = 3;
constexpr int kE  = 2400000;
constexpr int kB  = 4096;
constexpr int kN  = kNU + kNI;
constexpr int kH  = 32;       // gate hidden width
constexpr int kK  = 2 * kD;   // gate input width = 128
}

union AFrag { v16h v; _Float16 e[16]; };
union CFrag { v8f  v; float    f[8];  };

// ---------------------------------------------------------------- zero
__global__ void zero_kernel(float* __restrict__ p, int count) {
  int i = blockIdx.x * blockDim.x + threadIdx.x;
  int stride = gridDim.x * blockDim.x;
  for (; i < count; i += stride) p[i] = 0.0f;
}

// ---------------------------------------------------------------- SpMM (edge-parallel, one wave per edge)
__global__ void spmm_kernel(const int* __restrict__ rows,
                            const int* __restrict__ cols,
                            const float* __restrict__ vals,
                            const float* __restrict__ X,
                            float* __restrict__ out) {
  const int lane = threadIdx.x & 31;
  const int wavesPerBlock = blockDim.x >> 5;
  int e = blockIdx.x * wavesPerBlock + (threadIdx.x >> 5);
  const int stride = gridDim.x * wavesPerBlock;
  for (; e < kE; e += stride) {
    const int   r = rows[e];
    const int   c = cols[e];
    const float v = vals[e];
    const float2 x = *(const float2*)(X + (size_t)c * kD + lane * 2);
    float* dst = out + (size_t)r * kD + lane * 2;
    __hip_atomic_fetch_add(dst,     v * x.x, __ATOMIC_RELAXED, __HIP_MEMORY_SCOPE_AGENT);
    __hip_atomic_fetch_add(dst + 1, v * x.y, __ATOMIC_RELAXED, __HIP_MEMORY_SCOPE_AGENT);
  }
}

// ---------------------------------------------------------------- gate MLP (WMMA) + blend
// gate input per row: [XA_row (64) ‖ XB_row (64)]; output blend: g*XA + (1-g)*XO
template <int ACCUM>
__global__ __launch_bounds__(32)
void gate_fuse_kernel(const float* __restrict__ XA,
                      const float* __restrict__ XB,
                      const float* __restrict__ XO,
                      float* __restrict__ out,
                      float* __restrict__ accb,
                      const float* __restrict__ W1,   // [128][32]
                      const float* __restrict__ b1v,  // [32]
                      const float* __restrict__ W2,   // [32]
                      const float* __restrict__ b2v,  // [1]
                      int nrows) {
  __shared__ _Float16 s_w1t[kH][kK + 8];   // W1 transposed (N-major), padded vs bank conflicts
  __shared__ float    s_b1[kH];
  __shared__ float    s_w2[kH];
  __shared__ float    s_b2;
  __shared__ _Float16 s_x[16][kK];         // 16-row input tile, f16
  __shared__ float    s_alpha[16];

  const int lane = threadIdx.x;            // 0..31, one wave per block

  // stage layer weights once per block
  for (int i = lane; i < kK * kH; i += 32) {
    int n = i & (kH - 1);
    int k = i >> 5;
    s_w1t[n][k] = (_Float16)W1[k * kH + n];
  }
  s_b1[lane] = b1v[lane];
  s_w2[lane] = W2[lane];
  if (lane == 0) s_b2 = b2v[0];
  __syncthreads();

  const int h = lane >> 4;   // lane half
  const int m = lane & 15;   // A-frag row / B-frag column / C-frag N

  const int ntiles = (nrows + 15) >> 4;
  for (int tile = blockIdx.x; tile < ntiles; tile += gridDim.x) {
    // stage X tile (16 rows x 128 cols) into LDS as f16, coalesced
    for (int i = 0; i < 64; ++i) {
      int idx = i * 32 + lane;
      int row = idx >> 7;
      int c   = idx & 127;
      int rg  = tile * 16 + row;
      float val = 0.0f;
      if (rg < nrows)
        val = (c < kD) ? XA[(size_t)rg * kD + c] : XB[(size_t)rg * kD + (c - kD)];
      s_x[row][c] = (_Float16)val;
    }
    __syncthreads();

    // hidden = X @ W1 via 4 K-chunks x 2 N-tiles of v_wmma_f32_16x16x32_f16
    v8f acc0 = {};
    v8f acc1 = {};
    for (int c = 0; c < 4; ++c) {
      AFrag a;
#pragma unroll
      for (int e2 = 0; e2 < 16; ++e2) {
        int k = c * 32 + (e2 & 7) + h * 8 + ((e2 >> 3) << 4);  // ISA 16-bit A layout
        a.e[e2] = s_x[m][k];
      }
      AFrag b0, b1f;
#pragma unroll
      for (int e2 = 0; e2 < 16; ++e2) {
        int k = c * 32 + (e2 & 7) + h * 8 + ((e2 >> 3) << 4);  // ISA 16-bit B layout
        b0.e[e2]  = s_w1t[m][k];
        b1f.e[e2] = s_w1t[m + 16][k];
      }
      acc0 = __builtin_amdgcn_wmma_f32_16x16x32_f16(false, a.v, false, b0.v,
                                                    (short)0, acc0, false, false);
      acc1 = __builtin_amdgcn_wmma_f32_16x16x32_f16(false, a.v, false, b1f.v,
                                                    (short)0, acc1, false, false);
    }

    // epilogue: +b1, relu, dot with W2 (reduce across the 16 lanes of each half), +b2, sigmoid
    CFrag h0, h1;
    h0.v = acc0; h1.v = acc1;
    float partial[8];
#pragma unroll
    for (int v = 0; v < 8; ++v) {
      float x0 = fmaxf(h0.f[v] + s_b1[m],      0.0f) * s_w2[m];
      float x1 = fmaxf(h1.f[v] + s_b1[m + 16], 0.0f) * s_w2[m + 16];
      partial[v] = x0 + x1;
    }
#pragma unroll
    for (int off = 8; off >= 1; off >>= 1) {
#pragma unroll
      for (int v = 0; v < 8; ++v)
        partial[v] += __shfl_xor(partial[v], off, 16);
    }
    if (m == 0) {
#pragma unroll
      for (int v = 0; v < 8; ++v)
        s_alpha[h * 8 + v] = 1.0f / (1.0f + expf(-(partial[v] + s_b2)));  // rows M = 8h+v
    }
    __syncthreads();

    // blend in f32 and store; also update running layer-sum
    for (int r = 0; r < 16; ++r) {
      int rg = tile * 16 + r;
      if (rg >= nrows) break;
      float g = s_alpha[r];
      size_t base = (size_t)rg * kD + lane * 2;
#pragma unroll
      for (int j = 0; j < 2; ++j) {
        float fa = XA[base + j];
        float fo = XO[base + j];
        float fused = g * fa + (1.0f - g) * fo;
        out[base + j] = fused;
        if (ACCUM) accb[base + j] += fused;
        else       accb[base + j]  = fused;
      }
    }
    __syncthreads();
  }
}

// ---------------------------------------------------------------- users passthrough per layer
__global__ void copy_accum_kernel(const float* __restrict__ src,
                                  float* __restrict__ dst,
                                  float* __restrict__ accb,
                                  int count) {
  int i = blockIdx.x * blockDim.x + threadIdx.x;
  int stride = gridDim.x * blockDim.x;
  for (; i < count; i += stride) {
    float v = src[i];
    dst[i] = v;
    accb[i] += v;
  }
}

// ---------------------------------------------------------------- final B=4096 dot products
__global__ void dot_kernel(const float* __restrict__ accb,
                           const int* __restrict__ users,
                           const int* __restrict__ items,
                           float* __restrict__ outp) {
  const int lane = threadIdx.x & 31;
  const int wavesPerBlock = blockDim.x >> 5;
  int b = blockIdx.x * wavesPerBlock + (threadIdx.x >> 5);
  if (b >= kB) return;
  const int u  = users[b];
  const int it = items[b];
  const float2 xu = *(const float2*)(accb + (size_t)u * kD + lane * 2);
  const float2 xi = *(const float2*)(accb + ((size_t)(kNU + it)) * kD + lane * 2);
  float p = xu.x * xi.x + xu.y * xi.y;
#pragma unroll
  for (int off = 16; off >= 1; off >>= 1) p += __shfl_xor(p, off, 32);
  if (lane == 0) outp[b] = p * (1.0f / 16.0f);   // mean over 4 layers on both sides
}

// ---------------------------------------------------------------- launch
extern "C" void kernel_launch(void* const* d_in, const int* in_sizes, int n_in,
                              void* d_out, int out_size, void* d_ws, size_t ws_size,
                              hipStream_t stream) {
  (void)in_sizes; (void)n_in; (void)out_size; (void)ws_size;
  const float* emb_user  = (const float*)d_in[0];
  const float* emb_item  = (const float*)d_in[1];
  const float* sym_emb   = (const float*)d_in[2];
  const float* herb_emb  = (const float*)d_in[3];
  const float* gate_W1   = (const float*)d_in[4];
  const float* gate_b1   = (const float*)d_in[5];
  const float* gate_W2   = (const float*)d_in[6];
  const float* gate_b2   = (const float*)d_in[7];
  const float* base_vals = (const float*)d_in[8];
  const float* co_vals   = (const float*)d_in[9];
  const int*   users     = (const int*)d_in[10];
  const int*   items     = (const int*)d_in[11];
  const int*   base_rows = (const int*)d_in[12];
  const int*   base_cols = (const int*)d_in[13];
  const int*   co_rows   = (const int*)d_in[14];
  const int*   co_cols   = (const int*)d_in[15];
  float* out = (float*)d_out;

  const size_t embElems = (size_t)kN * kD;          // 9.6M floats
  float* all_emb  = (float*)d_ws;
  float* base_emb = all_emb + embElems;
  float* co_emb   = base_emb + embElems;
  float* accb     = co_emb + embElems;              // running sum of layer embeddings

  const int tilesU = (kNU + 15) / 16;
  const int tilesI = (kNI + 15) / 16;
  const int gU = tilesU < 2048 ? tilesU : 2048;
  const int gI = tilesI < 2048 ? tilesI : 2048;

  // layer 0: gated blend of (user|sym) and (item|herb); initializes all_emb and accb
  gate_fuse_kernel<0><<<gU, 32, 0, stream>>>(
      emb_user, sym_emb, sym_emb, all_emb, accb,
      gate_W1, gate_b1, gate_W2, gate_b2, kNU);
  gate_fuse_kernel<0><<<gI, 32, 0, stream>>>(
      emb_item, herb_emb, herb_emb,
      all_emb + (size_t)kNU * kD, accb + (size_t)kNU * kD,
      gate_W1, gate_b1, gate_W2, gate_b2, kNI);

  for (int layer = 1; layer <= kL; ++layer) {
    // base_emb and co_emb are contiguous in the workspace: zero both in one pass
    zero_kernel<<<4096, 256, 0, stream>>>(base_emb, (int)(2 * embElems));
    spmm_kernel<<<20000, 256, 0, stream>>>(base_rows, base_cols, base_vals, all_emb, base_emb);
    spmm_kernel<<<20000, 256, 0, stream>>>(co_rows, co_cols, co_vals, all_emb, co_emb);
    // users: passthrough of base_emb
    copy_accum_kernel<<<4096, 256, 0, stream>>>(base_emb, all_emb, accb, kNU * kD);
    // items: gate([base_items ‖ herb]) blending base_items with co_items
    gate_fuse_kernel<1><<<gI, 32, 0, stream>>>(
        base_emb + (size_t)kNU * kD, herb_emb, co_emb + (size_t)kNU * kD,
        all_emb + (size_t)kNU * kD, accb + (size_t)kNU * kD,
        gate_W1 + layer * (kK * kH), gate_b1 + layer * kH,
        gate_W2 + layer * kH, gate_b2 + layer, kNI);
  }

  dot_kernel<<<(kB + 7) / 8, 256, 0, stream>>>(accb, users, items, out);
}
